// CustomModel_51462298140705
// MI455X (gfx1250) — compile-verified, matrix operand order checked by reading, not run on hardware
//
#include <hip/hip_runtime.h>
#include <hip/hip_bf16.h>
#include <math.h>

#define NF 4096
#define ED 1152
#define NQ 16
#define CLIPSZ 8
#define NCLIPS (NF / CLIPSZ)
#define FEPS 1e-6f

typedef __attribute__((ext_vector_type(16))) __bf16 v16bf;
typedef __attribute__((ext_vector_type(8)))  __bf16 v8bf;
typedef __attribute__((ext_vector_type(8)))  float  v8f;
typedef __attribute__((ext_vector_type(4)))  unsigned u32x4;
typedef __attribute__((ext_vector_type(8)))  unsigned u32x8;

// Build a 16-element bf16 WMMA operand for one lane.
// 16-bit operand layout (A 16x32 / B 32x16): lanes 0-15 hold K=[0..8)+[16..24),
// lanes 16-31 hold K=[8..16)+[24..32).  Caller passes p = row_base + kc + half*8.
static __device__ __forceinline__ v16bf load_op16(const __bf16* p) {
  v8bf lo = *reinterpret_cast<const v8bf*>(p);
  v8bf hi = *reinterpret_cast<const v8bf*>(p + 16);
  return __builtin_shufflevector(lo, hi, 0,1,2,3,4,5,6,7,8,9,10,11,12,13,14,15);
}

static __device__ __forceinline__ v8f wmma_bf16(v16bf a, v16bf b, v8f c) {
  return __builtin_amdgcn_wmma_f32_16x16x32_bf16(false, a, false, b, (short)0, c,
                                                 false, false);
}

static __device__ __forceinline__ v8f v8f_zero() {
  v8f z = {0.f,0.f,0.f,0.f,0.f,0.f,0.f,0.f};
  return z;
}

// ---------------------------------------------------------------------------
// Tensor Data Mover: DMA a contiguous blob of bf16 (nelem elements, must fit
// a 16-bit tile_dim0) from global memory into LDS.  D# layout per
// CDNA5 ISA (group0 128b + group1 256b).
// Issue from a single wave; completion via s_wait_tensorcnt, then barrier.
// ---------------------------------------------------------------------------
static __device__ __forceinline__ void tdm_load_blob(const __bf16* g,
                                                     __bf16* lds_p,
                                                     unsigned nelem) {
  unsigned lds_off = (unsigned)(size_t)lds_p;          // LDS byte address
  unsigned long long ga = (unsigned long long)(size_t)g;
  u32x4 g0 = {
      1u,                                              // count=1, user desc
      lds_off,                                         // lds_addr
      (unsigned)(ga & 0xffffffffu),                    // global_addr[31:0]
      (unsigned)((ga >> 32) & 0x01ffffffu) | (2u << 30) // global_addr[56:32] | type=2
  };
  u32x8 g1 = {
      1u << 16,                                        // data_size=2B; mask=0
      (nelem & 0xffffu) << 16,                         // tensor_dim0[15:0]
      (nelem >> 16) | (1u << 16),                      // tensor_dim0[31:16] | tensor_dim1.lo=1
      (nelem & 0xffffu) << 16,                         // tensor_dim1.hi=0 | tile_dim0
      1u,                                              // tile_dim1=1, tile_dim2=0
      nelem,                                           // tensor_dim0_stride[31:0]
      0u,                                              // stride hi | dim1_stride lo
      0u                                               // dim1_stride hi
  };
  asm volatile("tensor_load_to_lds %0, %1" :: "s"(g0), "s"(g1) : "memory");
}

static __device__ float block_reduce128(float v, float* red) {
  int tid = threadIdx.x;
  red[tid] = v; __syncthreads();
  for (int s = 64; s > 0; s >>= 1) {
    if (tid < s) red[tid] += red[tid + s];
    __syncthreads();
  }
  float r = red[0]; __syncthreads();
  return r;
}

// ---------------------------------------------------------------------------
// Prep: per-frame norms; bf16 copies of frames (raw + eps-normalized) and V^T.
// ---------------------------------------------------------------------------
__global__ void k_prep_frames(const float* __restrict__ fe,
                              __bf16* __restrict__ fr_bf,
                              __bf16* __restrict__ nf_bf,
                              __bf16* __restrict__ vt_bf,
                              float* __restrict__ inv_fe,
                              float* __restrict__ inv_eps) {
  __shared__ float red[128];
  const int r = blockIdx.x;
  const float* row = fe + (size_t)r * ED;
  float ss = 0.f;
  for (int d = threadIdx.x; d < ED; d += 128) { float x = row[d]; ss += x * x; }
  ss = block_reduce128(ss, red);
  const float nrm = sqrtf(ss);
  const float ine = 1.f / (nrm + FEPS);   // eps-normalized (attention inputs)
  const float inn = 1.f / nrm;            // exact l2norm (fe_n)
  if (threadIdx.x == 0) { inv_fe[r] = inn; inv_eps[r] = ine; }
  for (int d = threadIdx.x; d < ED; d += 128) {
    float x = row[d];
    fr_bf[(size_t)r * ED + d] = (__bf16)x;
    nf_bf[(size_t)r * ED + d] = (__bf16)(x * ine);
    vt_bf[(size_t)d * NF + r] = (__bf16)x;   // V transposed for P@V B-operand
  }
}

__global__ void k_prep_text(const float* __restrict__ te, __bf16* __restrict__ te_bf) {
  __shared__ float red[128];
  const int r = blockIdx.x;
  const float* row = te + (size_t)r * ED;
  float ss = 0.f;
  for (int d = threadIdx.x; d < ED; d += 128) { float x = row[d]; ss += x * x; }
  ss = block_reduce128(ss, red);
  const float inv = rsqrtf(ss);
  for (int d = threadIdx.x; d < ED; d += 128)
    te_bf[(size_t)r * ED + d] = (__bf16)(row[d] * inv);
}

// ---------------------------------------------------------------------------
// Local clip attention (8x8 per clip). Softmax denominator cancels under the
// later l2norm, so we emit the unnormalized fused rows.
// ---------------------------------------------------------------------------
__global__ void k_local(const float* __restrict__ fe,
                        const float* __restrict__ inv_eps,
                        const float* __restrict__ tau_l,
                        __bf16* __restrict__ loc_bf) {
  __shared__ float FR[CLIPSZ * ED];     // 36 KB
  __shared__ float PART[64][4];
  __shared__ float SC[64];
  __shared__ float PR[64];
  __shared__ float INVE[CLIPSZ];
  const int c = blockIdx.x, tid = threadIdx.x;
  const float* base = fe + (size_t)c * CLIPSZ * ED;
  for (int i = tid; i < CLIPSZ * ED; i += 256) FR[i] = base[i];
  if (tid < CLIPSZ) INVE[tid] = inv_eps[c * CLIPSZ + tid];
  __syncthreads();
  const float invTL = 1.f / (__expf(tau_l[0]) * sqrtf((float)ED));
  {
    const int p = tid >> 2, q = tid & 3;
    const int i = p >> 3, j = p & 7;
    float s = 0.f;
    const int d0 = q * (ED / 4);
    for (int d = d0; d < d0 + ED / 4; ++d)
      s += FR[i * ED + d] * FR[j * ED + d];
    PART[p][q] = s;
  }
  __syncthreads();
  if (tid < 64) {
    const int i = tid >> 3, j = tid & 7;
    const float dot = PART[tid][0] + PART[tid][1] + PART[tid][2] + PART[tid][3];
    SC[tid] = dot * INVE[i] * INVE[j] * invTL;
  }
  __syncthreads();
  if (tid < CLIPSZ) {
    float m = -1e30f;
    for (int j = 0; j < CLIPSZ; ++j) m = fmaxf(m, SC[tid * 8 + j]);
    for (int j = 0; j < CLIPSZ; ++j) PR[tid * 8 + j] = __expf(SC[tid * 8 + j] - m);
  }
  __syncthreads();
  for (int d = tid; d < ED; d += 256) {
    #pragma unroll
    for (int i = 0; i < CLIPSZ; ++i) {
      float acc = 0.f;
      #pragma unroll
      for (int j = 0; j < CLIPSZ; ++j) acc += PR[i * 8 + j] * FR[j * ED + d];
      loc_bf[(size_t)(c * CLIPSZ + i) * ED + d] = (__bf16)acc;
    }
  }
}

// ---------------------------------------------------------------------------
// Global attention, flash-style single pass.  Scores are Gram dots of unit
// vectors, so exp(invTS*(dot-1)) is statically safe (row max = diagonal ~ 1),
// and the softmax denominator cancels under the later l2norm.
// Workgroup = 8 waves, one 16-query tile; O kept in registers (72 f32/lane).
// Q tile staged into LDS via the Tensor Data Mover.
// ---------------------------------------------------------------------------
__global__ void __launch_bounds__(256)
k_gattn(const __bf16* __restrict__ nf, const __bf16* __restrict__ vt,
        const float* __restrict__ tau_g, __bf16* __restrict__ glob_bf) {
  __shared__ __align__(16) __bf16 Qt[16 * ED];     // 36,864 B
  __shared__ __align__(16) __bf16 Pt[16 * 136];    // padded to avoid bank conflicts
  const int tid = threadIdx.x;
  const int wv = tid >> 5, lane = tid & 31, half = lane >> 4, lr = lane & 15;
  const int qb = blockIdx.x * 16;
  if (wv == 0) {
    tdm_load_blob(nf + (size_t)qb * ED, Qt, 16 * ED);
    __builtin_amdgcn_s_wait_tensorcnt(0);
  }
  __syncthreads();
  const float invTS = 1.f / (__expf(tau_g[0]) * sqrtf((float)ED));

  v8f acc[9];
  #pragma unroll
  for (int j = 0; j < 9; ++j) acc[j] = v8f_zero();

  const __bf16* Arow = &Qt[lr * ED];
  #pragma unroll 1
  for (int kb = 0; kb < NF; kb += 128) {
    // --- S tile: this wave computes 16 queries x 16 keys over K=1152 ---
    const int key = kb + wv * 16 + lr;
    const __bf16* Brow = nf + (size_t)key * ED;
    v8f c = v8f_zero();
    #pragma unroll 4
    for (int kc = 0; kc < ED; kc += 32) {
      v16bf a = load_op16(Arow + kc + half * 8);
      v16bf b = load_op16(Brow + kc + half * 8);
      c = wmma_bf16(a, b, c);
    }
    // --- P = exp(invTS*(dot-1)), bf16 into LDS (C layout -> row-major) ---
    #pragma unroll
    for (int v = 0; v < 8; ++v) {
      float p = __expf(invTS * (c[v] - 1.0f));
      Pt[(v + half * 8) * 136 + wv * 16 + lr] = (__bf16)p;
    }
    __syncthreads();
    // --- O += P(16x128) @ V(128xED); kk outer so A is loaded once per chunk ---
    #pragma unroll
    for (int kk = 0; kk < 128; kk += 32) {
      v16bf a = load_op16(&Pt[lr * 136 + kk + half * 8]);
      #pragma unroll
      for (int j = 0; j < 9; ++j) {
        const int dbase = 16 * (wv * 9 + j);
        const __bf16* Bv = vt + (size_t)(dbase + lr) * NF + kb + kk;
        acc[j] = wmma_bf16(a, load_op16(Bv + half * 8), acc[j]);
      }
    }
    __syncthreads();
    if (kb + 128 < NF)
      __builtin_prefetch(nf + (size_t)(kb + 128 + wv * 16 + lr) * ED, 0, 3);
  }
  // store unnormalized global_fused (norm applied later; cancels softmax Z)
  #pragma unroll
  for (int j = 0; j < 9; ++j) {
    const int dbase = 16 * (wv * 9 + j);
    #pragma unroll
    for (int v = 0; v < 8; ++v)
      glob_bf[(size_t)(qb + v + half * 8) * ED + dbase + lr] = (__bf16)acc[j][v];
  }
}

// ---------------------------------------------------------------------------
// Inverse L2 norms of loc/glob rows (eps = 0, per reference).
// ---------------------------------------------------------------------------
__global__ void k_rownorm(const __bf16* __restrict__ loc_bf,
                          const __bf16* __restrict__ glob_bf,
                          float* __restrict__ inv_loc,
                          float* __restrict__ inv_glob) {
  __shared__ float red[128];
  const int b = blockIdx.x;
  const bool is_loc = (b < NF);
  const int r = is_loc ? b : b - NF;
  const __bf16* src = is_loc ? loc_bf : glob_bf;
  float ss = 0.f;
  for (int d = threadIdx.x; d < ED; d += 128) {
    float x = (float)src[(size_t)r * ED + d];
    ss += x * x;
  }
  ss = block_reduce128(ss, red);
  if (threadIdx.x == 0) {
    float inv = rsqrtf(ss);
    if (is_loc) inv_loc[r] = inv; else inv_glob[r] = inv;
  }
}

// ---------------------------------------------------------------------------
// Logits: te_n @ {fe_n, loc_n, glob_n}.T via WMMA, then fused max over queries.
// One wave per 16-frame tile; te_n tile staged in LDS via TDM.
// ---------------------------------------------------------------------------
__global__ void __launch_bounds__(256)
k_logits(const __bf16* __restrict__ te_bf, const __bf16* __restrict__ fr_bf,
         const __bf16* __restrict__ loc_bf, const __bf16* __restrict__ glob_bf,
         const float* __restrict__ inv_fe, const float* __restrict__ inv_loc,
         const float* __restrict__ inv_glob,
         const float* __restrict__ logit_scale, const float* __restrict__ logit_bias,
         float* __restrict__ logits) {
  __shared__ __align__(16) __bf16 TE[16 * ED];
  const int tid = threadIdx.x;
  const int wv = tid >> 5, lane = tid & 31, half = lane >> 4, lr = lane & 15;
  if (wv == 0) {
    tdm_load_blob(te_bf, TE, 16 * ED);
    __builtin_amdgcn_s_wait_tensorcnt(0);
  }
  __syncthreads();
  const int fbase = (blockIdx.x * 8 + wv) * 16;
  v8f co = v8f_zero(), cl = v8f_zero(), cg = v8f_zero();
  const __bf16* Arow = &TE[lr * ED];
  const __bf16* Bo = fr_bf  + (size_t)(fbase + lr) * ED;
  const __bf16* Bl = loc_bf + (size_t)(fbase + lr) * ED;
  const __bf16* Bg = glob_bf + (size_t)(fbase + lr) * ED;
  #pragma unroll 2
  for (int kc = 0; kc < ED; kc += 32) {
    v16bf a = load_op16(Arow + kc + half * 8);
    co = wmma_bf16(a, load_op16(Bo + kc + half * 8), co);
    cl = wmma_bf16(a, load_op16(Bl + kc + half * 8), cl);
    cg = wmma_bf16(a, load_op16(Bg + kc + half * 8), cg);
  }
  const int f = fbase + lr;
  const float io = inv_fe[f], il = inv_loc[f], ig = inv_glob[f];
  const float ls = __expf(logit_scale[0]);
  const float bias = logit_bias[0];
  float m = -1e30f;
  #pragma unroll
  for (int v = 0; v < 8; ++v) {
    float fu = ls * (0.9f * co[v] * io + 0.05f * cl[v] * il + 0.05f * cg[v] * ig) + bias;
    m = fmaxf(m, fu);
  }
  m = fmaxf(m, __shfl_xor(m, 16, 32));   // combine the two 8-query halves
  if (half == 0) logits[f] = m;
}

// ---------------------------------------------------------------------------
// Softmax pooling over 8-frame clips + label-weighted log-sigmoid loss.
// ---------------------------------------------------------------------------
__global__ void k_pool_loss(const float* __restrict__ logits,
                            const float* __restrict__ labels,
                            float* __restrict__ out) {
  __shared__ float red[NCLIPS];
  const int c = threadIdx.x;   // 512 clips
  const float lab = labels[c * CLIPSZ];
  red[c] = lab; __syncthreads();
  for (int s = NCLIPS / 2; s > 0; s >>= 1) {
    if (c < s) red[c] += red[c + s];
    __syncthreads();
  }
  const float mean = red[0] / (float)NCLIPS;
  __syncthreads();
  float x[CLIPSZ]; float m = -1e30f;
  #pragma unroll
  for (int k = 0; k < CLIPSZ; ++k) { x[k] = logits[c * CLIPSZ + k]; m = fmaxf(m, x[k]); }
  float sw = 0.f, swx = 0.f;
  #pragma unroll
  for (int k = 0; k < CLIPSZ; ++k) { float w = __expf(x[k] - m); sw += w; swx += w * x[k]; }
  const float pooled = swx / sw;
  out[1 + c] = pooled;
  const float wgt = pooled * (lab - mean);
  const float lsg = fminf(wgt, 0.f) - log1pf(__expf(-fabsf(wgt)));
  red[c] = lsg; __syncthreads();
  for (int s = NCLIPS / 2; s > 0; s >>= 1) {
    if (c < s) red[c] += red[c + s];
    __syncthreads();
  }
  if (c == 0) out[0] = -red[0];
}

// ---------------------------------------------------------------------------
extern "C" void kernel_launch(void* const* d_in, const int* in_sizes, int n_in,
                              void* d_out, int out_size, void* d_ws, size_t ws_size,
                              hipStream_t stream) {
  (void)in_sizes; (void)n_in; (void)out_size; (void)ws_size;
  const float* fe     = (const float*)d_in[0];
  const float* te     = (const float*)d_in[1];
  const float* labels = (const float*)d_in[2];
  const float* tau_l  = (const float*)d_in[3];
  const float* tau_g  = (const float*)d_in[4];
  const float* lscale = (const float*)d_in[5];
  const float* lbias  = (const float*)d_in[6];
  float* out = (float*)d_out;

  char* ws = (char*)d_ws;
  size_t off = 0;
  auto carve = [&](size_t bytes) -> void* {
    off = (off + 255) & ~(size_t)255;
    void* p = ws + off;
    off += bytes;
    return p;
  };
  __bf16* fr_bf   = (__bf16*)carve((size_t)NF * ED * 2);
  __bf16* nf_bf   = (__bf16*)carve((size_t)NF * ED * 2);
  __bf16* vt_bf   = (__bf16*)carve((size_t)ED * NF * 2);
  __bf16* loc_bf  = (__bf16*)carve((size_t)NF * ED * 2);
  __bf16* glob_bf = (__bf16*)carve((size_t)NF * ED * 2);
  __bf16* te_bf   = (__bf16*)carve((size_t)NQ * ED * 2);
  float* inv_fe   = (float*)carve((size_t)NF * 4);
  float* inv_eps  = (float*)carve((size_t)NF * 4);
  float* inv_loc  = (float*)carve((size_t)NF * 4);
  float* inv_glob = (float*)carve((size_t)NF * 4);
  float* logits   = (float*)carve((size_t)NF * 4);

  k_prep_frames<<<NF, 128, 0, stream>>>(fe, fr_bf, nf_bf, vt_bf, inv_fe, inv_eps);
  k_prep_text<<<NQ, 128, 0, stream>>>(te, te_bf);
  k_local<<<NCLIPS, 256, 0, stream>>>(fe, inv_eps, tau_l, loc_bf);
  k_gattn<<<NF / 16, 256, 0, stream>>>(nf_bf, vt_bf, tau_g, glob_bf);
  k_rownorm<<<2 * NF, 128, 0, stream>>>(loc_bf, glob_bf, inv_loc, inv_glob);
  k_logits<<<NF / 16 / 8, 256, 0, stream>>>(te_bf, fr_bf, loc_bf, glob_bf,
                                            inv_fe, inv_loc, inv_glob,
                                            lscale, lbias, logits);
  k_pool_loss<<<1, NCLIPS, 0, stream>>>(logits, labels, out);
}
